// MultiHeadedAttention_51737176047655
// MI455X (gfx1250) — matrix operand order repaired and checked by reading, not compiled, not run on hardware
//
#include <hip/hip_runtime.h>
#include <hip/hip_bf16.h>
#include <math.h>

typedef __bf16 bf16_t;
typedef __attribute__((ext_vector_type(16))) __bf16 v16bf;
typedef __attribute__((ext_vector_type(8)))  float  v8f;

union Frag32 {
  v16bf  v;
  uint4  u[2];
  bf16_t h[16];
};

#define LDSB 40  // LDS row stride in halves (80B, multiple of 16B)

__device__ __forceinline__ v8f v8f_zero() {
  v8f z;
#pragma unroll
  for (int i = 0; i < 8; ++i) z[i] = 0.0f;
  return z;
}

__device__ __forceinline__ v8f wmma_bf16(v16bf a, v16bf b, v8f c) {
  // D = A(16x32 bf16) x B(32x16 bf16) + C(16x16 f32)
  return __builtin_amdgcn_wmma_f32_16x16x32_bf16(false, a, false, b, (short)0, c,
                                                 false, false);
}

__device__ __forceinline__ float redmax16(float x) {
#pragma unroll
  for (int off = 1; off < 16; off <<= 1) x = fmaxf(x, __shfl_xor(x, off, 32));
  return x;
}
__device__ __forceinline__ float redsum16(float x) {
#pragma unroll
  for (int off = 1; off < 16; off <<= 1) x += __shfl_xor(x, off, 32);
  return x;
}

__device__ __forceinline__ void cvt16f_to_bf16(Frag32& f, const float4* p) {
  float4 x0 = p[0], x1 = p[1], x2 = p[2], x3 = p[3];
  float v[16] = {x0.x, x0.y, x0.z, x0.w, x1.x, x1.y, x1.z, x1.w,
                 x2.x, x2.y, x2.z, x2.w, x3.x, x3.y, x3.z, x3.w};
#pragma unroll
  for (int u = 0; u < 16; ++u) f.h[u] = (bf16_t)v[u];
}

// Raw LDS byte offset of a __shared__ object (generic -> AS3 -> int).
__device__ __forceinline__ unsigned lds_off(const void* p) {
  return (unsigned)(unsigned long long)(
      (const __attribute__((address_space(3))) void*)p);
}

// ---------------------------------------------------------------------------
// GEMM: Out = A(MxK) * W(KxN f32) + bias, 128x128 tile, 8 waves (2x4), bf16 WMMA
// ---------------------------------------------------------------------------
template <bool A_BF16, bool OUT_HEADS>
__global__ __launch_bounds__(256) void gemm_kernel(
    const void* __restrict__ Aptr, const float* __restrict__ W,
    const float* __restrict__ bias, void* __restrict__ Out, int M, int K, int N,
    float scale) {
  __shared__ alignas(16) bf16_t As[128 * LDSB];
  __shared__ alignas(16) bf16_t Bs[128 * LDSB];

  const int tid = threadIdx.x;
  const int lane = tid & 31;
  const int wave = tid >> 5;
  const int wm = wave >> 2;  // 0..1
  const int wn = wave & 3;   // 0..3
  const int hs = lane >> 4;  // half-wave select
  const int l16 = lane & 15;
  const int m0 = blockIdx.x * 128;
  const int n0 = blockIdx.y * 128;

  v8f acc[4][2];
#pragma unroll
  for (int i = 0; i < 4; ++i)
#pragma unroll
    for (int j = 0; j < 2; ++j) acc[i][j] = v8f_zero();

  const int arow = tid >> 1;        // 0..127
  const int acol = (tid & 1) * 16;  // 0/16
  const int brow = tid >> 3;        // 0..31 (k)
  const int bcol = (tid & 7) * 16;  // 0..112 (n)

  for (int kk = 0; kk < K; kk += 32) {
    // prefetch next k-step tiles while this one is consumed
    if (kk + 32 < K) {
      if constexpr (A_BF16)
        __builtin_prefetch((const bf16_t*)Aptr + (size_t)(m0 + arow) * K + kk +
                               32 + acol, 0, 0);
      else
        __builtin_prefetch((const float*)Aptr + (size_t)(m0 + arow) * K + kk +
                               32 + acol, 0, 0);
      __builtin_prefetch(W + (size_t)(kk + 32 + brow) * N + n0 + bcol, 0, 0);
    }
    __syncthreads();
    // ---- A tile -> As[row][k] (bf16, row-major)
    {
      Frag32 fa;
      if constexpr (A_BF16) {
        const uint4* Ap = (const uint4*)((const bf16_t*)Aptr +
                                         (size_t)(m0 + arow) * K + kk + acol);
        fa.u[0] = Ap[0];
        fa.u[1] = Ap[1];
      } else {
        const float4* Ap = (const float4*)((const float*)Aptr +
                                           (size_t)(m0 + arow) * K + kk + acol);
        cvt16f_to_bf16(fa, Ap);
      }
      *(uint4*)&As[arow * LDSB + acol] = fa.u[0];
      *(uint4*)&As[arow * LDSB + acol + 8] = fa.u[1];
    }
    // ---- W tile -> Bs[n][k] (transposed, bf16)
    {
      const float4* Wp =
          (const float4*)(W + (size_t)(kk + brow) * N + n0 + bcol);
      float4 x0 = Wp[0], x1 = Wp[1], x2 = Wp[2], x3 = Wp[3];
      float v[16] = {x0.x, x0.y, x0.z, x0.w, x1.x, x1.y, x1.z, x1.w,
                     x2.x, x2.y, x2.z, x2.w, x3.x, x3.y, x3.z, x3.w};
#pragma unroll
      for (int u = 0; u < 16; ++u)
        Bs[(bcol + u) * LDSB + brow] = (bf16_t)v[u];
    }
    __syncthreads();

    Frag32 af[4], bfrag[2];
#pragma unroll
    for (int mt = 0; mt < 4; ++mt) {
      int r = wm * 64 + mt * 16 + l16;
      af[mt].u[0] = *(const uint4*)&As[r * LDSB + hs * 8];
      af[mt].u[1] = *(const uint4*)&As[r * LDSB + 16 + hs * 8];
    }
#pragma unroll
    for (int nt = 0; nt < 2; ++nt) {
      int c = wn * 32 + nt * 16 + l16;
      bfrag[nt].u[0] = *(const uint4*)&Bs[c * LDSB + hs * 8];
      bfrag[nt].u[1] = *(const uint4*)&Bs[c * LDSB + 16 + hs * 8];
    }
#pragma unroll
    for (int mt = 0; mt < 4; ++mt)
#pragma unroll
      for (int nt = 0; nt < 2; ++nt)
        acc[mt][nt] = wmma_bf16(af[mt].v, bfrag[nt].v, acc[mt][nt]);
  }

  // ---- epilogue
#pragma unroll
  for (int mt = 0; mt < 4; ++mt) {
#pragma unroll
    for (int nt = 0; nt < 2; ++nt) {
      int colg = n0 + wn * 32 + nt * 16 + l16;
      float bv = bias[colg];
#pragma unroll
      for (int r = 0; r < 8; ++r) {
        int rowg = m0 + wm * 64 + mt * 16 + r + 8 * hs;
        float val = (acc[mt][nt][r] + bv) * scale;
        if constexpr (OUT_HEADS) {
          int bb = rowg >> 10, ll = rowg & 1023;
          int hh = colg >> 6, hd = colg & 63;
          ((bf16_t*)Out)[((((size_t)bb * 16 + hh) << 10) + ll) * 64 + hd] =
              (bf16_t)val;
        } else {
          ((float*)Out)[(size_t)rowg * N + colg] = val;
        }
      }
    }
  }
}

// ---------------------------------------------------------------------------
// QE = Qh(bh,1024x64) x E[1024:2048]^T, tri-masked (m<=i), bf16 out [bh][i][m]
// ---------------------------------------------------------------------------
__global__ __launch_bounds__(256) void qe_kernel(const bf16_t* __restrict__ Qh,
                                                 const float* __restrict__ E,
                                                 bf16_t* __restrict__ QEm) {
  const int tid = threadIdx.x, lane = tid & 31, wave = tid >> 5;
  const int wm = wave >> 2, wn = wave & 3;
  const int hs = lane >> 4, l16 = lane & 15;
  const int i0 = blockIdx.x * 128, m0 = blockIdx.y * 128;
  const int bh = blockIdx.z;
  const bf16_t* Qb = Qh + ((size_t)bh << 16);

  v8f acc[4][2];
#pragma unroll
  for (int i = 0; i < 4; ++i)
#pragma unroll
    for (int j = 0; j < 2; ++j) acc[i][j] = v8f_zero();

#pragma unroll
  for (int ks = 0; ks < 2; ++ks) {
    Frag32 bfr[2];
#pragma unroll
    for (int nt = 0; nt < 2; ++nt) {
      int m = m0 + wn * 32 + nt * 16 + l16;
      const float4* Ep =
          (const float4*)(E + (size_t)(1024 + m) * 64 + ks * 32 + hs * 16);
      cvt16f_to_bf16(bfr[nt], Ep);
    }
#pragma unroll
    for (int mt = 0; mt < 4; ++mt) {
      int row = i0 + wm * 64 + mt * 16 + l16;
      Frag32 af;
      const bf16_t* p = Qb + (size_t)row * 64 + ks * 32 + hs * 16;
      af.u[0] = *(const uint4*)p;
      af.u[1] = *(const uint4*)(p + 8);
#pragma unroll
      for (int nt = 0; nt < 2; ++nt)
        acc[mt][nt] = wmma_bf16(af.v, bfr[nt].v, acc[mt][nt]);
    }
  }

#pragma unroll
  for (int mt = 0; mt < 4; ++mt) {
#pragma unroll
    for (int nt = 0; nt < 2; ++nt) {
#pragma unroll
      for (int r = 0; r < 8; ++r) {
        int i = i0 + wm * 64 + mt * 16 + r + 8 * hs;
        int m = m0 + wn * 32 + nt * 16 + l16;
        float v = (m <= i) ? acc[mt][nt][r] : 0.0f;
        QEm[((size_t)bh << 20) + ((size_t)i << 10) + m] = (bf16_t)v;
      }
    }
  }
}

// ---------------------------------------------------------------------------
// Flash attention: block = 128 queries (8 waves x 16 rows) for one (b,h),
// streaming 64-key blocks with online softmax. V tile is staged into LDS with
// CDNA5 async global->LDS copies (ASYNCcnt), then transposed LDS->LDS into the
// WMMA B-fragment layout. Srel gathered from tri-masked QE.
// ---------------------------------------------------------------------------
__global__ __launch_bounds__(256) void attn_kernel(
    const bf16_t* __restrict__ Qh, const bf16_t* __restrict__ Kh,
    const bf16_t* __restrict__ Vh, const bf16_t* __restrict__ QEm,
    const unsigned char* __restrict__ maskp, bf16_t* __restrict__ ctx) {
  __shared__ alignas(16) bf16_t Vraw[64 * 64];    // async landing, row-major
  __shared__ alignas(16) bf16_t Vt[64 * 72];      // V^T tile [hd][key]
  __shared__ alignas(16) bf16_t Pw[8 * 16 * 72];  // per-wave P staging
  __shared__ unsigned char smask[64];

  const int tid = threadIdx.x, lane = tid & 31, wave = tid >> 5;
  const int hs = lane >> 4, l16 = lane & 15;
  const int bh = blockIdx.y;
  const int b = bh >> 4, h = bh & 15;
  const int i0 = blockIdx.x * 128;
  const int qbase = i0 + wave * 16;

  const bf16_t* Qb = Qh + ((size_t)bh << 16);
  const bf16_t* Kb = Kh + ((size_t)bh << 16);
  const bf16_t* Vb = Vh + ((size_t)bh << 16);
  const bf16_t* QEb = QEm + ((size_t)bh << 20);

  // Q fragments (A-layout) straight from global: M = lane%16
  Frag32 qf[2];
#pragma unroll
  for (int ks = 0; ks < 2; ++ks) {
    const bf16_t* p = Qb + (size_t)(qbase + l16) * 64 + ks * 32 + hs * 8;
    qf[ks].u[0] = *(const uint4*)p;
    qf[ks].u[1] = *(const uint4*)(p + 16);
  }

  v8f cacc[4];
#pragma unroll
  for (int nt = 0; nt < 4; ++nt) cacc[nt] = v8f_zero();
  float mrow[8], lrow[8];
#pragma unroll
  for (int r = 0; r < 8; ++r) {
    mrow[r] = -__builtin_inff();
    lrow[r] = 0.0f;
  }

  bf16_t* P = &Pw[wave * 16 * 72];
  const int vrow = tid >> 2;         // 0..63
  const int vcol = (tid & 3) * 16;   // 0,16,32,48 (halves)

  for (int j0 = 0; j0 < 1024; j0 += 64) {
    __syncthreads();  // prev-iter Vraw/Vt consumers done

    // ---- async global->LDS copy of V tile (no VGPR staging, ASYNCcnt)
    {
      const bf16_t* src = Vb + (size_t)(j0 + vrow) * 64 + vcol;
      unsigned dst = lds_off(&Vraw[vrow * 64 + vcol]);
      asm volatile(
          "global_load_async_to_lds_b128 %0, %2, off\n\t"
          "global_load_async_to_lds_b128 %1, %3, off"
          :
          : "v"(dst), "v"(dst + 16u), "v"(src), "v"(src + 8)
          : "memory");
      if (tid < 64) smask[tid] = maskp[(size_t)b * 1024 + j0 + tid];
      if (j0 + 64 < 1024)  // prefetch next K block
        __builtin_prefetch(Kb + (size_t)(j0 + 64 + vrow) * 64, 0, 0);
      asm volatile("s_wait_asynccnt 0x0" ::: "memory");
    }
    __syncthreads();  // Vraw complete (all waves waited their async ops)

    // ---- transpose Vraw -> Vt (B-fragment friendly [hd][key] layout)
    {
      Frag32 fv;
      fv.u[0] = *(const uint4*)&Vraw[vrow * 64 + vcol];
      fv.u[1] = *(const uint4*)&Vraw[vrow * 64 + vcol + 8];
#pragma unroll
      for (int u = 0; u < 16; ++u) Vt[(vcol + u) * 72 + vrow] = fv.h[u];
    }
    __syncthreads();

    // ---- S = Q * K^T (K rows are already B-fragment shaped in global)
    v8f s[4];
#pragma unroll
    for (int nt = 0; nt < 4; ++nt) {
      s[nt] = v8f_zero();
#pragma unroll
      for (int ks = 0; ks < 2; ++ks) {
        Frag32 kf;
        const bf16_t* p =
            Kb + (size_t)(j0 + nt * 16 + l16) * 64 + ks * 32 + hs * 8;
        kf.u[0] = *(const uint4*)p;
        kf.u[1] = *(const uint4*)(p + 16);
        s[nt] = wmma_bf16(qf[ks].v, kf.v, s[nt]);
      }
    }

    // ---- + Srel (gather from tri-masked QE), key mask
#pragma unroll
    for (int r = 0; r < 8; ++r) {
      const int i = qbase + r + 8 * hs;
      const int jb = j0 + l16;  // j = jb + nt*16
      const bf16_t* prow = QEb + ((size_t)i << 10) + (size_t)(jb - i + 1023);
#pragma unroll
      for (int nt = 0; nt < 4; ++nt) {
        float sv = s[nt][r];
        if (jb + nt * 16 <= i) sv += (float)prow[nt * 16];
        sv = smask[nt * 16 + l16] ? sv : -__builtin_inff();
        s[nt][r] = sv;
      }
    }

    // ---- online softmax (per-row reductions across 16 lanes)
#pragma unroll
    for (int r = 0; r < 8; ++r) {
      float tm = fmaxf(fmaxf(s[0][r], s[1][r]), fmaxf(s[2][r], s[3][r]));
      tm = redmax16(tm);
      float newm = fmaxf(mrow[r], tm);
      float alpha = __expf(mrow[r] - newm);
      mrow[r] = newm;
      float rs = 0.0f;
#pragma unroll
      for (int nt = 0; nt < 4; ++nt) {
        float p = __expf(s[nt][r] - newm);
        s[nt][r] = p;
        rs += p;
      }
      rs = redsum16(rs);
      lrow[r] = lrow[r] * alpha + rs;
#pragma unroll
      for (int nt = 0; nt < 4; ++nt) cacc[nt][r] *= alpha;
    }

    // ---- P (C-layout f32) -> per-wave LDS bf16, reload as A fragments
#pragma unroll
    for (int nt = 0; nt < 4; ++nt)
#pragma unroll
      for (int r = 0; r < 8; ++r)
        P[(r + 8 * hs) * 72 + nt * 16 + l16] = (bf16_t)s[nt][r];

    __builtin_amdgcn_wave_barrier();
    asm volatile("s_wait_dscnt 0x0" ::: "memory");

    Frag32 pf[2];
#pragma unroll
    for (int ks = 0; ks < 2; ++ks) {
      const bf16_t* p = &P[l16 * 72 + ks * 32 + hs * 8];
      pf[ks].u[0] = *(const uint4*)p;
      pf[ks].u[1] = *(const uint4*)(p + 16);
    }

    // ---- ctx += P * V
#pragma unroll
    for (int nt = 0; nt < 4; ++nt) {
#pragma unroll
      for (int ks = 0; ks < 2; ++ks) {
        Frag32 vf;
        const bf16_t* p = &Vt[(nt * 16 + l16) * 72 + ks * 32 + hs * 8];
        vf.u[0] = *(const uint4*)p;
        vf.u[1] = *(const uint4*)(p + 16);
        cacc[nt] = wmma_bf16(pf[ks].v, vf.v, cacc[nt]);
      }
    }
  }

  // ---- normalize, write ctx in [b][l][h*64+hd] flat layout
#pragma unroll
  for (int nt = 0; nt < 4; ++nt) {
    int col = h * 64 + nt * 16 + l16;
#pragma unroll
    for (int r = 0; r < 8; ++r) {
      int i = qbase + r + 8 * hs;
      float inv = 1.0f / lrow[r];
      ctx[(((size_t)b << 10) + i) * 1024 + col] = (bf16_t)(cacc[nt][r] * inv);
    }
  }
}

// ---------------------------------------------------------------------------
extern "C" void kernel_launch(void* const* d_in, const int* in_sizes, int n_in,
                              void* d_out, int out_size, void* d_ws,
                              size_t ws_size, hipStream_t stream) {
  (void)in_sizes;
  (void)n_in;
  (void)out_size;
  (void)ws_size;
  const float* k = (const float*)d_in[0];
  const float* v = (const float*)d_in[1];
  const float* q = (const float*)d_in[2];
  const unsigned char* mask = (const unsigned char*)d_in[3];
  const float* E = (const float*)d_in[4];
  const float* Wk = (const float*)d_in[5];
  const float* bk = (const float*)d_in[6];
  const float* Wv = (const float*)d_in[7];
  const float* bv = (const float*)d_in[8];
  const float* Wq = (const float*)d_in[9];
  const float* bq = (const float*)d_in[10];
  const float* Wo = (const float*)d_in[11];
  const float* bo = (const float*)d_in[12];

  char* ws = (char*)d_ws;
  bf16_t* Qh = (bf16_t*)(ws);                        //  8 MB [b][h][l][64]
  bf16_t* Kh = (bf16_t*)(ws + ((size_t)8 << 20));    //  8 MB
  bf16_t* Vh = (bf16_t*)(ws + ((size_t)16 << 20));   //  8 MB
  bf16_t* ctx = (bf16_t*)(ws + ((size_t)24 << 20));  //  8 MB [b][l][1024]
  bf16_t* QEm = (bf16_t*)(ws + ((size_t)32 << 20));  // 128 MB [bh][i][m]

  dim3 blk(256);
  dim3 gG(32, 8);  // 4096/128 x 1024/128

  const float inv_sqrt_hd = 0.125f;  // 1/sqrt(64), folded into Q projection
  gemm_kernel<false, true><<<gG, blk, 0, stream>>>(k, Wk, bk, Kh, 4096, 1024,
                                                   1024, 1.0f);
  gemm_kernel<false, true><<<gG, blk, 0, stream>>>(v, Wv, bv, Vh, 4096, 1024,
                                                   1024, 1.0f);
  gemm_kernel<false, true><<<gG, blk, 0, stream>>>(q, Wq, bq, Qh, 4096, 1024,
                                                   1024, inv_sqrt_hd);

  qe_kernel<<<dim3(8, 8, 64), blk, 0, stream>>>(Qh, E, QEm);

  attn_kernel<<<dim3(8, 64), blk, 0, stream>>>(Qh, Kh, Vh, QEm, mask, ctx);

  gemm_kernel<true, false><<<gG, blk, 0, stream>>>(ctx, Wo, bo, d_out, 4096,
                                                   1024, 1024, 1.0f);
}
